// PIScoreNet_51110110822714
// MI455X (gfx1250) — matrix operand order repaired and checked by reading, not compiled, not run on hardware
//
// PIScoreNet for MI455X (gfx1250), compile-only WMMA implementation, round 3.
//
// Changes vs round 2 (driven by disasm):
//  - Activations now use the gfx1250 hardware transcendental path instead of
//    libm expansions: __expf -> v_exp_f32 (TRANS, co-executes with the WMMA
//    pipe), reciprocal via __builtin_amdgcn_rcpf -> v_rcp_f32. GELU uses the
//    tanh form folded into a single sigmoid: 0.5x(1+tanh z) = x*sigma(2z).
//    This collapses the ~25-VALU-per-element IEEE expf range reduction that
//    dominated round 2's valu count; accuracy loss is below the f16-WMMA
//    quantization already present in the data path.
//
// Retained design:
//  - Weights staged in LDS in WMMA B-fragment order -> two aligned v8h loads
//    (2x ds_load_b128) per fragment; linear_kernel keeps B fragments in
//    registers across its persistent tile loop.
//  - Fused LDS-resident f16 WMMA chains (v_wmma_f32_16x16x32_f16, f32
//    accumulate) for every linear layer; gather/scatter via
//    global_atomic_add_f32 into the 192MB L2 (workload is gather-bound:
//    1.6M-edge EGNN + 500K-pair heads vs ~50 GFLOP of skinny GEMMs).

#include <hip/hip_runtime.h>

typedef __attribute__((ext_vector_type(16))) _Float16 v16h;
typedef __attribute__((ext_vector_type(8)))  _Float16 v8h;
typedef __attribute__((ext_vector_type(2)))  _Float16 v2h;
typedef __attribute__((ext_vector_type(8)))  float    v8f;
typedef __attribute__((ext_vector_type(4)))  float    v4f;
typedef __attribute__((ext_vector_type(2)))  float    v2f;

#define N_RES  10000
#define N_ATM  100000
#define N_HIGH 30000
#define E_RES  160000
#define E_ATM  1600000
#define E_HIGH 480000
#define N_PAIR 500000

enum { ACT_NONE = 0, ACT_ELU = 1, ACT_SILU = 2, ACT_GELU = 3 };

__device__ __forceinline__ float fast_sigmoid(float v) {
  return __builtin_amdgcn_rcpf(1.f + __expf(-v));   // v_exp_f32 + v_rcp_f32
}
__device__ __forceinline__ float act_apply(float v, int act) {
  if (act == ACT_ELU)  return v > 0.f ? v : (__expf(v) - 1.f);
  if (act == ACT_SILU) return v * fast_sigmoid(v);
  if (act == ACT_GELU) {
    // 0.5*v*(1+tanh(z)) == v*sigmoid(2z),  z = sqrt(2/pi)*(v + 0.044715 v^3)
    const float z = 0.7978845608f * (v + 0.044715f * v * v * v);
    return v * fast_sigmoid(2.f * z);
  }
  return v;
}

// ---------------------------------------------------------------------------
// Weight staging: global f32 (KINxMOUT row-major) -> LDS f16 in B-fragment
// order:  Ws[((k>>5)*MP + n)*32 + (k&31)].  A lane's fragment (k 8h..8h+7 and
// 16+8h..23+8h of one column, h = lane/16) is then two aligned 16-byte loads.
// ---------------------------------------------------------------------------
template<int KP, int MP>
__device__ __forceinline__ void stage_weights(const float* __restrict__ W,
                                              int KIN, int MOUT,
                                              _Float16* __restrict__ Ws,
                                              int tid, int nthr) {
  for (int i = tid; i < KP * MP; i += nthr) {
    const int k = i / MP, n = i - k * MP;
    const float v = (k < KIN && n < MOUT) ? W[k * MOUT + n] : 0.f;
    Ws[((k >> 5) * MP + n) * 32 + (k & 31)] = (_Float16)v;
  }
}

// B fragment: lanes 0-15 hold K {0..7, 16..23}, lanes 16-31 hold K {8..15, 24..31} of column n.
template<int MP>
__device__ __forceinline__ v16h load_bfrag(const _Float16* __restrict__ Ws,
                                           int kt, int ncol, int lane) {
  const int kb = (lane >> 4) * 8;
  const v8h b0 = *(const v8h*)&Ws[(kt * MP + ncol) * 32 + kb];
  const v8h b1 = *(const v8h*)&Ws[(kt * MP + ncol) * 32 + kb + 16];
  return __builtin_shufflevector(b0, b1, 0, 1, 2, 3, 4, 5, 6, 7,
                                 8, 9, 10, 11, 12, 13, 14, 15);
}

// A fragment from a row-major 16xKP LDS tile (CDNA5 ISA 7.12.2 16-bit A):
// lanes 0-15: row m, K {c0..c0+7, c0+16..c0+23} with c0=ko; lanes 16-31: +8.
template<int KP>
__device__ __forceinline__ v16h load_afrag(const _Float16* __restrict__ A,
                                           int ko, int lane) {
  const int m  = lane & 15;
  const int c0 = ko + (lane >> 4) * 8;
  const v8h a0 = *(const v8h*)&A[m * KP + c0];
  const v8h a1 = *(const v8h*)&A[m * KP + c0 + 16];
  return __builtin_shufflevector(a0, a1, 0, 1, 2, 3, 4, 5, 6, 7,
                                 8, 9, 10, 11, 12, 13, 14, 15);
}

// GEMM with B fragments loaded from swizzled LDS each k-step.
template<int KP, int MP>
__device__ __forceinline__ void wave_gemm(const _Float16* __restrict__ A,
                                          const _Float16* __restrict__ Ws,
                                          int lane, v8f* acc) {
  const int nl = lane & 15;
#pragma unroll
  for (int kt = 0; kt < KP / 32; ++kt) {
    const v16h a = load_afrag<KP>(A, kt * 32, lane);
#pragma unroll
    for (int nt = 0; nt < MP / 16; ++nt) {
      const v16h b = load_bfrag<MP>(Ws, kt, nt * 16 + nl, lane);
      acc[nt] = __builtin_amdgcn_wmma_f32_16x16x32_f16(
          false, a, false, b, (short)0, acc[nt], false, false);
    }
  }
}

// GEMM with register-resident B fragments (hoisted out of a tile loop).
template<int KP, int MP>
__device__ __forceinline__ void wave_gemm_reg(const _Float16* __restrict__ A,
                                              const v16h* __restrict__ Breg,
                                              int lane, v8f* acc) {
#pragma unroll
  for (int kt = 0; kt < KP / 32; ++kt) {
    const v16h a = load_afrag<KP>(A, kt * 32, lane);
#pragma unroll
    for (int nt = 0; nt < MP / 16; ++nt) {
      acc[nt] = __builtin_amdgcn_wmma_f32_16x16x32_f16(
          false, a, false, Breg[kt * (MP / 16) + nt], (short)0, acc[nt],
          false, false);
    }
  }
}

// Store a 16x32 accumulator tile (bias + activation) to LDS f16 and/or f32.
__device__ __forceinline__ void store_tile_16x32(const v8f* acc, const float* bias,
                                                 int lane, int act,
                                                 _Float16* d16, float* d32) {
  const int nl = lane & 15;
  const int hb = (lane >> 4) ? 8 : 0;
#pragma unroll
  for (int t = 0; t < 2; ++t) {
    const int col = t * 16 + nl;
    const float bb = bias[col];
#pragma unroll
    for (int r = 0; r < 8; ++r) {
      const float v = act_apply(acc[t][r] + bb, act);
      if (d16) d16[(r + hb) * 32 + col] = (_Float16)v;
      if (d32) d32[(r + hb) * 32 + col] = v;
    }
  }
}

// ---------------------------------------------------------------------------
// Generic linear layer: Y = act(X[:, xoff:xoff+KIN] @ W + b).
// Persistent grid; B fragments live in registers for the whole tile loop.
// ---------------------------------------------------------------------------
template<int KP, int MP, int ACT>
__global__ __launch_bounds__(256) void linear_kernel(
    const float* __restrict__ X, int ldx, int xoff, int KIN,
    const float* __restrict__ W, const float* __restrict__ Bb, int MOUT,
    float* __restrict__ Y, int ldy, int n) {
  __shared__ alignas(32) _Float16 Wl[KP * MP];
  __shared__ float Bl[MP];
  __shared__ alignas(16) _Float16 Al[8][16 * KP];
  const int tid = threadIdx.x, lane = tid & 31, wv = tid >> 5;
  stage_weights<KP, MP>(W, KIN, MOUT, Wl, tid, 256);
  for (int i = tid; i < MP; i += 256) Bl[i] = (i < MOUT) ? Bb[i] : 0.f;
  __syncthreads();
  const int nl = lane & 15;
  const int hb = (lane >> 4) ? 8 : 0;
  v16h Breg[(KP / 32) * (MP / 16)];
#pragma unroll
  for (int kt = 0; kt < KP / 32; ++kt)
#pragma unroll
    for (int nt = 0; nt < MP / 16; ++nt)
      Breg[kt * (MP / 16) + nt] = load_bfrag<MP>(Wl, kt, nt * 16 + nl, lane);
  _Float16* A = Al[wv];
  const int ntiles = (n + 15) >> 4;
  for (int tile = blockIdx.x * 8 + wv; tile < ntiles; tile += gridDim.x * 8) {
    const int r0 = tile * 16;
    if (r0 + 128 < n) __builtin_prefetch(&X[(size_t)(r0 + 128) * ldx + xoff], 0, 1);
    for (int i = lane; i < 16 * (KP / 2); i += 32) {
      const int r = i / (KP / 2);
      const int k2 = (i - r * (KP / 2)) * 2;
      const int row = r0 + r;
      v2h p = {};
      if (row < n) {
        const size_t base = (size_t)row * ldx + xoff;
        if (k2 < KIN)     p[0] = (_Float16)X[base + k2];
        if (k2 + 1 < KIN) p[1] = (_Float16)X[base + k2 + 1];
      }
      *(v2h*)&A[r * KP + k2] = p;
    }
    v8f zero = {};
    v8f acc[MP / 16];
#pragma unroll
    for (int t = 0; t < MP / 16; ++t) acc[t] = zero;
    wave_gemm_reg<KP, MP>(A, Breg, lane, acc);
#pragma unroll
    for (int t = 0; t < MP / 16; ++t) {
      const int col = t * 16 + nl;
      if (col >= MOUT) continue;
      const float bb = Bl[col];
#pragma unroll
      for (int r = 0; r < 8; ++r) {
        const int row = r0 + r + hb;
        if (row < n) Y[(size_t)row * ldy + col] = act_apply(acc[t][r] + bb, ACT);
      }
    }
  }
}

// ---------------------------------------------------------------------------
// EGNN fused edge kernel (persistent): builds [h_row|h_col|d2|eattr] tiles,
// e1(96->32 silu) -> e2(32->32 silu)=m -> c1(32->32 silu) -> dot c2, then
// atomic scatter of coord update and messages. 128 threads = 4 waves.
// ---------------------------------------------------------------------------
__global__ __launch_bounds__(128) void egnn_edge_kernel(
    const float* __restrict__ h, const float* __restrict__ x,
    const int* __restrict__ eidx, const float* __restrict__ ea,
    const float* __restrict__ e1w, const float* __restrict__ e1b,
    const float* __restrict__ e2w, const float* __restrict__ e2b,
    const float* __restrict__ c1w, const float* __restrict__ c1b,
    const float* __restrict__ c2w,
    float* __restrict__ xacc, float* __restrict__ agg, int nE) {
  __shared__ alignas(32) _Float16 We1[96 * 32];
  __shared__ alignas(32) _Float16 We2[32 * 32];
  __shared__ alignas(32) _Float16 Wc1[32 * 32];
  __shared__ float Be1[32], Be2[32], Bc1[32];
  __shared__ alignas(16) float C2[32];
  __shared__ alignas(16) _Float16 Al[4][16 * 96];
  __shared__ alignas(16) _Float16 Tl[4][16 * 32];
  __shared__ alignas(16) float    Mf[4][16 * 32];
  __shared__ alignas(16) float    Cf[4][16 * 32];
  __shared__ float Df[4][16 * 3];
  __shared__ int   Ri[4][16];
  const int tid = threadIdx.x, lane = tid & 31, wv = tid >> 5;
  stage_weights<96, 32>(e1w, 67, 32, We1, tid, 128);
  stage_weights<32, 32>(e2w, 32, 32, We2, tid, 128);
  stage_weights<32, 32>(c1w, 32, 32, Wc1, tid, 128);
  for (int i = tid; i < 32; i += 128) {
    Be1[i] = e1b[i]; Be2[i] = e2b[i]; Bc1[i] = c1b[i]; C2[i] = c2w[i];
  }
  __syncthreads();
  _Float16* A = Al[wv];
  const int ntiles = (nE + 15) >> 4;
  for (int tile = blockIdx.x * 4 + wv; tile < ntiles; tile += gridDim.x * 4) {
    const int e0 = tile * 16;
    // geometry + edge attrs (lanes 0..15, one edge each)
    if (lane < 16) {
      const int e = e0 + lane;
      float dx = 0.f, dy = 0.f, dz = 0.f, d2 = 0.f, a0 = 0.f, a1 = 0.f;
      int r = 0;
      if (e < nE) {
        r = eidx[e];
        const int c = eidx[nE + e];
        dx = x[r * 3 + 0] - x[c * 3 + 0];
        dy = x[r * 3 + 1] - x[c * 3 + 1];
        dz = x[r * 3 + 2] - x[c * 3 + 2];
        d2 = dx * dx + dy * dy + dz * dz;
        const v2f av = *(const v2f*)&ea[(size_t)e * 2];
        a0 = av[0]; a1 = av[1];
      }
      Ri[wv][lane] = (e < nE) ? r : -1;
      Df[wv][lane * 3 + 0] = dx;
      Df[wv][lane * 3 + 1] = dy;
      Df[wv][lane * 3 + 2] = dz;
      A[lane * 96 + 64] = (_Float16)d2;
      A[lane * 96 + 65] = (_Float16)a0;
      A[lane * 96 + 66] = (_Float16)a1;
#pragma unroll
      for (int k = 68; k < 96; k += 2) *(v2h*)&A[lane * 96 + k] = (v2h){};
      A[lane * 96 + 67] = (_Float16)0.f;
    }
    // endpoint feature gather (paired loads/stores, rows are 8B aligned)
    for (int i = lane; i < 16 * 16; i += 32) {
      const int e = i >> 4;
      const int ch = (i & 15) * 2;
      const int ee = e0 + e;
      v2h p0 = {}, p1 = {};
      if (ee < nE) {
        const v2f hv = *(const v2f*)&h[(size_t)eidx[ee] * 32 + ch];
        const v2f cv = *(const v2f*)&h[(size_t)eidx[nE + ee] * 32 + ch];
        p0[0] = (_Float16)hv[0]; p0[1] = (_Float16)hv[1];
        p1[0] = (_Float16)cv[0]; p1[1] = (_Float16)cv[1];
      }
      *(v2h*)&A[e * 96 + ch]      = p0;
      *(v2h*)&A[e * 96 + 32 + ch] = p1;
    }
    v8f zero = {};
    v8f acc[2];
    acc[0] = zero; acc[1] = zero;
    wave_gemm<96, 32>(A, We1, lane, acc);
    store_tile_16x32(acc, Be1, lane, ACT_SILU, Tl[wv], nullptr);
    acc[0] = zero; acc[1] = zero;
    wave_gemm<32, 32>(Tl[wv], We2, lane, acc);
    store_tile_16x32(acc, Be2, lane, ACT_SILU, Tl[wv], Mf[wv]);   // m
    acc[0] = zero; acc[1] = zero;
    wave_gemm<32, 32>(Tl[wv], Wc1, lane, acc);
    store_tile_16x32(acc, Bc1, lane, ACT_SILU, nullptr, Cf[wv]);
    if (lane < 16) {
      const int e = e0 + lane;
      if (e < nE) {
        float cc = 0.f;
#pragma unroll
        for (int k = 0; k < 32; k += 4) {
          const v4f cv = *(const v4f*)&Cf[wv][lane * 32 + k];
          const v4f wv4 = *(const v4f*)&C2[k];
          cc += cv[0] * wv4[0] + cv[1] * wv4[1] + cv[2] * wv4[2] + cv[3] * wv4[3];
        }
        const int r = Ri[wv][lane];
        atomicAdd(&xacc[r * 3 + 0], Df[wv][lane * 3 + 0] * cc);
        atomicAdd(&xacc[r * 3 + 1], Df[wv][lane * 3 + 1] * cc);
        atomicAdd(&xacc[r * 3 + 2], Df[wv][lane * 3 + 2] * cc);
      }
    }
#pragma unroll 1
    for (int e = 0; e < 16; ++e) {
      const int ee = e0 + e;
      if (ee < nE)
        atomicAdd(&agg[(size_t)Ri[wv][e] * 32 + lane], Mf[wv][e * 32 + lane]);
    }
  }
}

// ---------------------------------------------------------------------------
// EGNN node update (persistent): x += xacc/clip(cnt,1); h += n2(silu(n1([h|agg]))).
// ---------------------------------------------------------------------------
__global__ __launch_bounds__(256) void egnn_node_kernel(
    float* __restrict__ h, float* __restrict__ x,
    const float* __restrict__ agg, const float* __restrict__ xacc,
    const float* __restrict__ cnt,
    const float* __restrict__ n1w, const float* __restrict__ n1b,
    const float* __restrict__ n2w, const float* __restrict__ n2b, int n) {
  __shared__ alignas(32) _Float16 W1[64 * 32];
  __shared__ alignas(32) _Float16 W2[32 * 32];
  __shared__ float B1[32], B2[32];
  __shared__ alignas(16) _Float16 Al[8][16 * 64];
  __shared__ alignas(16) _Float16 Tl[8][16 * 32];
  const int tid = threadIdx.x, lane = tid & 31, wv = tid >> 5;
  stage_weights<64, 32>(n1w, 64, 32, W1, tid, 256);
  stage_weights<32, 32>(n2w, 32, 32, W2, tid, 256);
  for (int i = tid; i < 32; i += 256) { B1[i] = n1b[i]; B2[i] = n2b[i]; }
  __syncthreads();
  const int nl = lane & 15;
  const int hb = (lane >> 4) ? 8 : 0;
  _Float16* A = Al[wv];
  const int ntiles = (n + 15) >> 4;
  for (int tile = blockIdx.x * 8 + wv; tile < ntiles; tile += gridDim.x * 8) {
    const int r0 = tile * 16;
    for (int i = lane; i < 48; i += 32) {
      const int r = i / 3, d = i - r * 3;
      const int row = r0 + r;
      if (row < n) {
        const float cv = fmaxf(cnt[row], 1.f);
        x[row * 3 + d] += xacc[row * 3 + d] / cv;
      }
    }
    for (int i = lane; i < 16 * 16; i += 32) {
      const int r = i >> 4;
      const int ch = (i & 15) * 2;
      const int row = r0 + r;
      v2h p0 = {}, p1 = {};
      if (row < n) {
        const v2f hv = *(const v2f*)&h[(size_t)row * 32 + ch];
        const v2f av = *(const v2f*)&agg[(size_t)row * 32 + ch];
        p0[0] = (_Float16)hv[0]; p0[1] = (_Float16)hv[1];
        p1[0] = (_Float16)av[0]; p1[1] = (_Float16)av[1];
      }
      *(v2h*)&A[r * 64 + ch]      = p0;
      *(v2h*)&A[r * 64 + 32 + ch] = p1;
    }
    v8f zero = {};
    v8f acc[2]; acc[0] = zero; acc[1] = zero;
    wave_gemm<64, 32>(A, W1, lane, acc);
    store_tile_16x32(acc, B1, lane, ACT_SILU, Tl[wv], nullptr);
    acc[0] = zero; acc[1] = zero;
    wave_gemm<32, 32>(Tl[wv], W2, lane, acc);
#pragma unroll
    for (int t = 0; t < 2; ++t) {
      const int col = t * 16 + nl;
      const float bb = B2[col];
#pragma unroll
      for (int r = 0; r < 8; ++r) {
        const int row = r0 + r + hb;
        if (row < n) h[(size_t)row * 32 + col] += acc[t][r] + bb;
      }
    }
  }
}

// ---------------------------------------------------------------------------
// Fused pair-head MLP3 (persistent): [h[i0]|h[i1]] -> 64->32 gelu -> 32->32
// gelu -> MOUT (1: per-lane dot, 20: third WMMA).
// ---------------------------------------------------------------------------
template<int MOUT>
__global__ __launch_bounds__(256) void pair_mlp3_kernel(
    const float* __restrict__ hn, const int* __restrict__ pidx,
    const float* w1, const float* b1, const float* w2, const float* b2,
    const float* w3, const float* b3, float* __restrict__ out, int np) {
  __shared__ alignas(32) _Float16 W1[64 * 32];
  __shared__ alignas(32) _Float16 W2[32 * 32];
  __shared__ alignas(32) _Float16 W3[32 * 32];
  __shared__ float B1[32], B2[32], B3[32];
  __shared__ alignas(16) float W3f[32];
  __shared__ alignas(16) _Float16 Al[8][16 * 64];
  __shared__ alignas(16) _Float16 Tl[8][16 * 32];
  __shared__ alignas(16) float    Cf[8][16 * 32];
  const int tid = threadIdx.x, lane = tid & 31, wv = tid >> 5;
  stage_weights<64, 32>(w1, 64, 32, W1, tid, 256);
  stage_weights<32, 32>(w2, 32, 32, W2, tid, 256);
  stage_weights<32, 32>(w3, 32, MOUT, W3, tid, 256);
  for (int i = tid; i < 32; i += 256) {
    B1[i] = b1[i]; B2[i] = b2[i];
    B3[i]  = (i < MOUT) ? b3[i] : 0.f;
    W3f[i] = w3[i * MOUT];
  }
  __syncthreads();
  const int nl = lane & 15;
  const int hb = (lane >> 4) ? 8 : 0;
  _Float16* A = Al[wv];
  const int ntiles = (np + 15) >> 4;
  for (int tile = blockIdx.x * 8 + wv; tile < ntiles; tile += gridDim.x * 8) {
    const int e0 = tile * 16;
    for (int i = lane; i < 16 * 16; i += 32) {
      const int e = i >> 4;
      const int ch = (i & 15) * 2;
      const int ee = e0 + e;
      v2h p0 = {}, p1 = {};
      if (ee < np) {
        const v2f hv = *(const v2f*)&hn[(size_t)pidx[ee] * 32 + ch];
        const v2f cv = *(const v2f*)&hn[(size_t)pidx[np + ee] * 32 + ch];
        p0[0] = (_Float16)hv[0]; p0[1] = (_Float16)hv[1];
        p1[0] = (_Float16)cv[0]; p1[1] = (_Float16)cv[1];
      }
      *(v2h*)&A[e * 64 + ch]      = p0;
      *(v2h*)&A[e * 64 + 32 + ch] = p1;
    }
    v8f zero = {};
    v8f acc[2]; acc[0] = zero; acc[1] = zero;
    wave_gemm<64, 32>(A, W1, lane, acc);
    store_tile_16x32(acc, B1, lane, ACT_GELU, Tl[wv], nullptr);
    acc[0] = zero; acc[1] = zero;
    wave_gemm<32, 32>(Tl[wv], W2, lane, acc);
    if constexpr (MOUT == 1) {
      store_tile_16x32(acc, B2, lane, ACT_GELU, nullptr, Cf[wv]);
      if (lane < 16) {
        const int row = e0 + lane;
        if (row < np) {
          float s = B3[0];
#pragma unroll
          for (int k = 0; k < 32; k += 4) {
            const v4f cv = *(const v4f*)&Cf[wv][lane * 32 + k];
            const v4f wv4 = *(const v4f*)&W3f[k];
            s += cv[0] * wv4[0] + cv[1] * wv4[1] + cv[2] * wv4[2] + cv[3] * wv4[3];
          }
          out[row] = s;
        }
      }
    } else {
      store_tile_16x32(acc, B2, lane, ACT_GELU, Tl[wv], nullptr);
      acc[0] = zero; acc[1] = zero;
      wave_gemm<32, 32>(Tl[wv], W3, lane, acc);
#pragma unroll
      for (int t = 0; t < 2; ++t) {
        const int col = t * 16 + nl;
        if (col >= MOUT) continue;
#pragma unroll
        for (int r = 0; r < 8; ++r) {
          const int row = e0 + r + hb;
          if (row < np) out[(size_t)row * MOUT + col] = acc[t][r] + B3[col];
        }
      }
    }
  }
}

// Fused node MLP3 (pa heads, persistent): 32->32 gelu -> 32->32 gelu -> 1.
__global__ __launch_bounds__(256) void node_mlp3_kernel(
    const float* __restrict__ X,
    const float* w1, const float* b1, const float* w2, const float* b2,
    const float* w3, const float* b3, float* __restrict__ out, int n) {
  __shared__ alignas(32) _Float16 W1[32 * 32];
  __shared__ alignas(32) _Float16 W2[32 * 32];
  __shared__ float B1[32], B2[32];
  __shared__ alignas(16) float W3[32];
  __shared__ alignas(16) _Float16 Al[8][16 * 32];
  __shared__ alignas(16) _Float16 Tl[8][16 * 32];
  __shared__ alignas(16) float    Cf[8][16 * 32];
  const int tid = threadIdx.x, lane = tid & 31, wv = tid >> 5;
  stage_weights<32, 32>(w1, 32, 32, W1, tid, 256);
  stage_weights<32, 32>(w2, 32, 32, W2, tid, 256);
  for (int i = tid; i < 32; i += 256) { B1[i] = b1[i]; B2[i] = b2[i]; W3[i] = w3[i]; }
  __syncthreads();
  _Float16* A = Al[wv];
  const int ntiles = (n + 15) >> 4;
  for (int tile = blockIdx.x * 8 + wv; tile < ntiles; tile += gridDim.x * 8) {
    const int r0 = tile * 16;
    for (int i = lane; i < 16 * 16; i += 32) {
      const int r = i >> 4;
      const int ch = (i & 15) * 2;
      const int row = r0 + r;
      v2h p = {};
      if (row < n) {
        const v2f hv = *(const v2f*)&X[(size_t)row * 32 + ch];
        p[0] = (_Float16)hv[0]; p[1] = (_Float16)hv[1];
      }
      *(v2h*)&A[r * 32 + ch] = p;
    }
    v8f zero = {};
    v8f acc[2]; acc[0] = zero; acc[1] = zero;
    wave_gemm<32, 32>(A, W1, lane, acc);
    store_tile_16x32(acc, B1, lane, ACT_GELU, Tl[wv], nullptr);
    acc[0] = zero; acc[1] = zero;
    wave_gemm<32, 32>(Tl[wv], W2, lane, acc);
    store_tile_16x32(acc, B2, lane, ACT_GELU, nullptr, Cf[wv]);
    if (lane < 16) {
      const int row = r0 + lane;
      if (row < n) {
        float s = b3[0];
#pragma unroll
        for (int k = 0; k < 32; k += 4) {
          const v4f cv = *(const v4f*)&Cf[wv][lane * 32 + k];
          const v4f wv4 = *(const v4f*)&W3[k];
          s += cv[0] * wv4[0] + cv[1] * wv4[1] + cv[2] * wv4[2] + cv[3] * wv4[3];
        }
        out[row] = s;
      }
    }
  }
}

// ------------------------------- small utilities ---------------------------
__global__ void zero_kernel(float* __restrict__ p, int n) {
  const int i = blockIdx.x * blockDim.x + threadIdx.x;
  if (i < n) p[i] = 0.f;
}
__global__ void copy_kernel(float* __restrict__ d, const float* __restrict__ s, int n) {
  const int i = blockIdx.x * blockDim.x + threadIdx.x;
  if (i < n) d[i] = s[i];
}
__global__ void count_kernel(float* __restrict__ cnt, const int* __restrict__ eidx, int nE) {
  const int i = blockIdx.x * blockDim.x + threadIdx.x;
  if (i < nE) atomicAdd(&cnt[eidx[i]], 1.f);
}
__global__ void gather_rows_kernel(float* __restrict__ dst, const float* __restrict__ src,
                                   const int* __restrict__ idx, int ncol, int n) {
  const int i = blockIdx.x * blockDim.x + threadIdx.x;
  if (i >= n * ncol) return;
  const int r = i / ncol, c = i - r * ncol;
  dst[i] = src[(size_t)idx[r] * ncol + c];
}
__global__ void concat2_kernel(float* __restrict__ dst, const float* __restrict__ a,
                               const float* __restrict__ b, int ca, int cb, int n) {
  const int i = blockIdx.x * blockDim.x + threadIdx.x;
  const int ct = ca + cb;
  if (i >= n * ct) return;
  const int r = i / ct, c = i - r * ct;
  dst[i] = (c < ca) ? a[(size_t)r * ca + c] : b[(size_t)r * cb + (c - ca)];
}
// proj_in = [atm_x[:, :1] | h00(15) | h01(15) | atm_x[:, 87:151]]  (95 cols)
__global__ void proj_build_kernel(const float* __restrict__ ax, const float* __restrict__ h00,
                                  const float* __restrict__ h01, float* __restrict__ out, int n) {
  const int i = blockIdx.x * blockDim.x + threadIdx.x;
  if (i >= n * 95) return;
  const int r = i / 95, c = i - r * 95;
  float v;
  if (c == 0)       v = ax[(size_t)r * 151];
  else if (c < 16)  v = h00[(size_t)r * 15 + (c - 1)];
  else if (c < 31)  v = h01[(size_t)r * 15 + (c - 16)];
  else              v = ax[(size_t)r * 151 + 87 + (c - 31)];
  out[i] = v;
}
// pool layout: [0:32) lig sums, [32:64) rec sums, [64] lig cnt, [65] rec cnt
__global__ void pool_accum_kernel(const float* __restrict__ hn, const int* __restrict__ mask,
                                  float* __restrict__ pool, int n) {
  const int i = blockIdx.x * blockDim.x + threadIdx.x;
  if (i >= n * 32) return;
  const int r = i >> 5, c = i & 31;
  const int m = mask[r] > 0;
  atomicAdd(&pool[(m ? 0 : 32) + c], hn[i]);
  if (c == 0) atomicAdd(&pool[64 + (m ? 0 : 1)], 1.f);
}
// Final feature-weight heads (tiny: ~2.2K MACs; single thread).
__global__ void fw_kernel(const float* __restrict__ pool,
                          const float* f1w, const float* f1b,
                          const float* f2w, const float* f2b,
                          const float* g1w, const float* g1b,
                          const float* g2w, const float* g2b,
                          float* __restrict__ out_f, float* __restrict__ out_g) {
  if (threadIdx.x != 0 || blockIdx.x != 0) return;
  float ctx[64];
  const float lc = fmaxf(pool[64], 1.f);
  const float rc = fmaxf(pool[65], 1.f);
  for (int k = 0; k < 32; ++k) { ctx[k] = pool[k] / lc; ctx[32 + k] = pool[32 + k] / rc; }
  const float* W1[2] = {f1w, g1w};
  const float* B1[2] = {f1b, g1b};
  const float* W2[2] = {f2w, g2w};
  const float* B2[2] = {f2b, g2b};
  float* O[2] = {out_f, out_g};
  for (int hd = 0; hd < 2; ++hd) {
    float h1[16];
    for (int j = 0; j < 16; ++j) {
      float s = B1[hd][j];
      for (int k = 0; k < 64; ++k) s += ctx[k] * W1[hd][k * 16 + j];
      h1[j] = act_apply(s, ACT_GELU);
    }
    float lg[5], mx = -1e30f;
    for (int j = 0; j < 5; ++j) {
      float s = B2[hd][j];
      for (int k = 0; k < 16; ++k) s += h1[k] * W2[hd][k * 5 + j];
      lg[j] = s; mx = fmaxf(mx, s);
    }
    float se = 0.f;
    for (int j = 0; j < 5; ++j) { lg[j] = __expf(lg[j] - mx); se += lg[j]; }
    for (int j = 0; j < 5; ++j) O[hd][j] = lg[j] / se;
    O[hd][5] = 0.1f;
  }
}

// ------------------------------- host side ---------------------------------
static inline unsigned cdiv(long a, long b) { return (unsigned)((a + b - 1) / b); }
static inline unsigned capg(long rows, long rpb, unsigned cap) {
  const unsigned g = cdiv(rows, rpb);
  return g < cap ? g : cap;
}

struct EgnnW {
  const float *inw, *inb, *outw, *outb;
  const float *e1w[2], *e1b[2], *e2w[2], *e2b[2];
  const float *c1w[2], *c1b[2], *c2[2];
  const float *n1w[2], *n1b[2], *n2w[2], *n2b[2];
};

static void run_egnn(hipStream_t st, const float* h0, int in_nf, float* xbuf,
                     const int* eidx, const float* eattr, int nE, int n,
                     const EgnnW& W, float* hbuf, float* hout,
                     float* cnt, float* xacc, float* agg) {
  if (in_nf == 64)
    linear_kernel<64, 32, ACT_NONE><<<capg(n, 128, 512), 256, 0, st>>>(
        h0, 64, 0, 64, W.inw, W.inb, 32, hbuf, 32, n);
  else
    linear_kernel<32, 32, ACT_NONE><<<capg(n, 128, 512), 256, 0, st>>>(
        h0, 32, 0, 32, W.inw, W.inb, 32, hbuf, 32, n);
  zero_kernel<<<cdiv(n, 256), 256, 0, st>>>(cnt, n);
  count_kernel<<<cdiv(nE, 256), 256, 0, st>>>(cnt, eidx, nE);
  for (int l = 0; l < 2; ++l) {
    zero_kernel<<<cdiv(3L * n, 256), 256, 0, st>>>(xacc, 3 * n);
    zero_kernel<<<cdiv(32L * n, 256), 256, 0, st>>>(agg, 32 * n);
    egnn_edge_kernel<<<capg(nE, 64, 2048), 128, 0, st>>>(
        hbuf, xbuf, eidx, eattr, W.e1w[l], W.e1b[l], W.e2w[l], W.e2b[l],
        W.c1w[l], W.c1b[l], W.c2[l], xacc, agg, nE);
    egnn_node_kernel<<<capg(n, 128, 512), 256, 0, st>>>(
        hbuf, xbuf, agg, xacc, cnt, W.n1w[l], W.n1b[l], W.n2w[l], W.n2b[l], n);
  }
  linear_kernel<32, 32, ACT_NONE><<<capg(n, 128, 512), 256, 0, st>>>(
      hbuf, 32, 0, 32, W.outw, W.outb, 32, hout, 32, n);
}

extern "C" void kernel_launch(void* const* d_in, const int* in_sizes, int n_in,
                              void* d_out, int out_size, void* d_ws, size_t ws_size,
                              hipStream_t stream) {
  (void)in_sizes; (void)n_in; (void)out_size; (void)ws_size;
  const float* res_x    = (const float*)d_in[0];
  const float* res_xyz  = (const float*)d_in[1];
  const int*   res_ei   = (const int*)d_in[2];
  const float* res_ea   = (const float*)d_in[3];
  const float* atm_x    = (const float*)d_in[4];
  const float* atm_xyz  = (const float*)d_in[5];
  const int*   atm_ei   = (const int*)d_in[6];
  const float* atm_ea   = (const float*)d_in[7];
  const int*   high_ei  = (const int*)d_in[8];
  const float* high_ea  = (const float*)d_in[9];
  const int*   res_idx  = (const int*)d_in[10];
  const int*   high_idx = (const int*)d_in[11];
  const int*   pair_i   = (const int*)d_in[12];
  const int*   lig_mask = (const int*)d_in[13];
  // params pytree flattened in JAX sorted-dict order:
  //   0-5 apol(l1b,l1w,l2b,l2w,l3b,l3w) | 6-31 atm_egnn(inb,inw, 2x[c1b,c1w,c2,
  //   e1b,e1w,e2b,e2w,n1b,n1w,n2b,n2w], outb,outw) | 32-37 dist | 38-41 fw_fnat
  //   | 42-45 fw_lddt | 46-51 hbond | 52-77 high_egnn | 78-79 linear_res(b,w)
  //   | 80-89 p00,p00b,p01,p01b,p1 (b,w each) | 90-95 pa | 96-101 pa_high
  //   | 102-107 pol | 108-133 res_egnn | 134-135 res_lin1 | 136-137 res_lin2
  auto P = [&](int i) { return (const float*)d_in[14 + i]; };

  auto fill_egnn = [&](EgnnW& W, int b) {
    W.inb = P(b); W.inw = P(b + 1);
    for (int l = 0; l < 2; ++l) {
      const int o = b + 2 + 11 * l;
      W.c1b[l] = P(o + 0); W.c1w[l] = P(o + 1); W.c2[l] = P(o + 2);
      W.e1b[l] = P(o + 3); W.e1w[l] = P(o + 4);
      W.e2b[l] = P(o + 5); W.e2w[l] = P(o + 6);
      W.n1b[l] = P(o + 7); W.n1w[l] = P(o + 8);
      W.n2b[l] = P(o + 9); W.n2w[l] = P(o + 10);
    }
    W.outb = P(b + 24); W.outw = P(b + 25);
  };
  EgnnW Wres, Watm, Whigh;
  fill_egnn(Watm, 6);
  fill_egnn(Whigh, 52);
  fill_egnn(Wres, 108);

  float* wsf = (float*)d_ws;
  size_t cur = 0;
  auto alloc = [&](size_t nf) { float* p = wsf + cur; cur += nf; return p; };
  float* t0      = alloc((size_t)N_RES * 35);
  float* hres0   = alloc((size_t)N_RES * 32);
  float* res_xw  = alloc((size_t)N_RES * 3);
  float* hres_h  = alloc((size_t)N_RES * 32);
  float* hres_o  = alloc((size_t)N_RES * 32);
  float* cnt_r   = alloc(N_RES);
  float* xacc_r  = alloc((size_t)N_RES * 3);
  float* agg_r   = alloc((size_t)N_RES * 32);
  float* tA      = alloc((size_t)N_ATM * 15);
  float* h00     = alloc((size_t)N_ATM * 15);
  float* tB      = alloc((size_t)N_ATM * 15);
  float* h01     = alloc((size_t)N_ATM * 15);
  float* proj_in = alloc((size_t)N_ATM * 95);
  float* h_proj  = alloc((size_t)N_ATM * 32);
  float* res_at  = alloc((size_t)N_ATM * 32);
  float* hatm_in = alloc((size_t)N_ATM * 64);
  float* atm_xw  = alloc((size_t)N_ATM * 3);
  float* hatm_h  = alloc((size_t)N_ATM * 32);
  float* hatm_o  = alloc((size_t)N_ATM * 32);
  float* cnt_a   = alloc(N_ATM);
  float* xacc_a  = alloc((size_t)N_ATM * 3);
  float* agg_a   = alloc((size_t)N_ATM * 32);
  float* hh_in   = alloc((size_t)N_HIGH * 32);
  float* high_xw = alloc((size_t)N_HIGH * 3);
  float* hh_h    = alloc((size_t)N_HIGH * 32);
  float* hh_o    = alloc((size_t)N_HIGH * 32);
  float* cnt_h   = alloc(N_HIGH);
  float* xacc_h  = alloc((size_t)N_HIGH * 3);
  float* agg_h   = alloc((size_t)N_HIGH * 32);
  float* node_in = alloc((size_t)N_ATM * 64);
  float* h_node  = alloc((size_t)N_ATM * 32);
  float* pool    = alloc(66);

  // ---- residue branch ----
  linear_kernel<64, 48, ACT_ELU><<<capg(N_RES, 128, 512), 256, 0, stream>>>(
      res_x, 35, 0, 35, P(135), P(134), 35, t0, 35, N_RES);
  linear_kernel<64, 32, ACT_NONE><<<capg(N_RES, 128, 512), 256, 0, stream>>>(
      t0, 35, 0, 35, P(137), P(136), 32, hres0, 32, N_RES);
  copy_kernel<<<cdiv(3 * N_RES, 256), 256, 0, stream>>>(res_xw, res_xyz, 3 * N_RES);
  run_egnn(stream, hres0, 32, res_xw, res_ei, res_ea, E_RES, N_RES, Wres,
           hres_h, hres_o, cnt_r, xacc_r, agg_r);

  // ---- atom projection ----
  linear_kernel<32, 16, ACT_ELU><<<capg(N_ATM, 128, 512), 256, 0, stream>>>(
      atm_x, 151, 1, 21, P(81), P(80), 15, tA, 15, N_ATM);
  linear_kernel<32, 16, ACT_NONE><<<capg(N_ATM, 128, 512), 256, 0, stream>>>(
      tA, 15, 0, 15, P(83), P(82), 15, h00, 15, N_ATM);
  linear_kernel<96, 16, ACT_ELU><<<capg(N_ATM, 128, 512), 256, 0, stream>>>(
      atm_x, 151, 22, 65, P(85), P(84), 15, tB, 15, N_ATM);
  linear_kernel<32, 16, ACT_NONE><<<capg(N_ATM, 128, 512), 256, 0, stream>>>(
      tB, 15, 0, 15, P(87), P(86), 15, h01, 15, N_ATM);
  proj_build_kernel<<<cdiv((long)N_ATM * 95, 256), 256, 0, stream>>>(
      atm_x, h00, h01, proj_in, N_ATM);
  linear_kernel<96, 32, ACT_ELU><<<capg(N_ATM, 128, 512), 256, 0, stream>>>(
      proj_in, 95, 0, 95, P(89), P(88), 32, h_proj, 32, N_ATM);

  // ---- atom EGNN ----
  gather_rows_kernel<<<cdiv((long)N_ATM * 32, 256), 256, 0, stream>>>(
      res_at, hres_o, res_idx, 32, N_ATM);
  concat2_kernel<<<cdiv((long)N_ATM * 64, 256), 256, 0, stream>>>(
      hatm_in, res_at, h_proj, 32, 32, N_ATM);
  copy_kernel<<<cdiv(3 * N_ATM, 256), 256, 0, stream>>>(atm_xw, atm_xyz, 3 * N_ATM);
  run_egnn(stream, hatm_in, 64, atm_xw, atm_ei, atm_ea, E_ATM, N_ATM, Watm,
           hatm_h, hatm_o, cnt_a, xacc_a, agg_a);

  // ---- high-level EGNN ----
  gather_rows_kernel<<<cdiv((long)N_HIGH * 32, 256), 256, 0, stream>>>(
      hh_in, hatm_o, high_idx, 32, N_HIGH);
  gather_rows_kernel<<<cdiv((long)N_HIGH * 3, 256), 256, 0, stream>>>(
      high_xw, atm_xyz, high_idx, 3, N_HIGH);
  run_egnn(stream, hh_in, 32, high_xw, high_ei, high_ea, E_HIGH, N_HIGH, Whigh,
           hh_h, hh_o, cnt_h, xacc_h, agg_h);

  // ---- node features ----
  concat2_kernel<<<cdiv((long)N_ATM * 64, 256), 256, 0, stream>>>(
      node_in, res_at, hatm_o, 32, 32, N_ATM);
  linear_kernel<64, 32, ACT_ELU><<<capg(N_ATM, 128, 512), 256, 0, stream>>>(
      node_in, 64, 0, 64, P(79), P(78), 32, h_node, 32, N_ATM);

  // ---- output heads (d_out: hbond|pol|apol|dist|pa|pa_high|fw_fnat|fw_lddt) ----
  float* out = (float*)d_out;
  pair_mlp3_kernel<1><<<capg(N_PAIR, 128, 1024), 256, 0, stream>>>(
      h_node, pair_i, P(47), P(46), P(49), P(48), P(51), P(50), out + 0, N_PAIR);
  pair_mlp3_kernel<1><<<capg(N_PAIR, 128, 1024), 256, 0, stream>>>(
      h_node, pair_i, P(103), P(102), P(105), P(104), P(107), P(106),
      out + 500000, N_PAIR);
  pair_mlp3_kernel<1><<<capg(N_PAIR, 128, 1024), 256, 0, stream>>>(
      h_node, pair_i, P(1), P(0), P(3), P(2), P(5), P(4), out + 1000000, N_PAIR);
  pair_mlp3_kernel<20><<<capg(N_PAIR, 128, 1024), 256, 0, stream>>>(
      h_node, pair_i, P(33), P(32), P(35), P(34), P(37), P(36),
      out + 1500000, N_PAIR);
  node_mlp3_kernel<<<capg(N_ATM, 128, 512), 256, 0, stream>>>(
      h_node, P(91), P(90), P(93), P(92), P(95), P(94), out + 11500000, N_ATM);
  node_mlp3_kernel<<<capg(N_HIGH, 128, 512), 256, 0, stream>>>(
      hh_o, P(97), P(96), P(99), P(98), P(101), P(100), out + 11600000, N_HIGH);

  zero_kernel<<<1, 128, 0, stream>>>(pool, 66);
  pool_accum_kernel<<<cdiv((long)N_ATM * 32, 256), 256, 0, stream>>>(
      h_node, lig_mask, pool, N_ATM);
  fw_kernel<<<1, 32, 0, stream>>>(pool, P(39), P(38), P(41), P(40),
                                  P(43), P(42), P(45), P(44),
                                  out + 11630000, out + 11630006);
}
